// VRNNlm_33973191311978
// MI455X (gfx1250) — matrix operand order repaired and checked by reading, not compiled, not run on hardware
//
#include <hip/hip_runtime.h>
#include <hip/hip_bf16.h>

// ---------------------------------------------------------------------------
// VRNN language model on MI455X (gfx1250, wave32, WMMA).
//  * bf16 weights in workspace, v_wmma_f32_16x16x32_bf16 for all linears
//  * generation loss fused: WMMA GEMM + online log-sum-exp (logits never
//    hit HBM); B-fragment loads batched into one clause to overlap the
//    XDL pipe with outstanding global loads
//  * __launch_bounds__(256,1) so the 8 A-frags + 8 B-frags stay register-
//    resident (round 2 spilled A to scratch inside the 157-iteration loop)
// ---------------------------------------------------------------------------

typedef __bf16 bf16_t;
typedef __attribute__((ext_vector_type(16))) __bf16 v16bf;
typedef __attribute__((ext_vector_type(8)))  __bf16 v8bf;
typedef __attribute__((ext_vector_type(8)))  float  v8f;

#define VSZ 10000
#define KCLS 4
#define XD 256
#define HD 512
#define ZD 128
#define NB 64
#define TT 256
#define RTOT (TT*NB)   // 16384 rows for generation loss

union Frag { v16bf v; v8bf h[2]; };

// ---------------------------------------------------------------------------
// Generic fused linear: out = act(A[M,K](bf16) @ W[N,K]^T(bf16) + bias)
// One wave computes one 16x16 output tile; 2-deep software pipeline on K.
// ACT: 0=none 1=relu 2=sigmoid 3=softplus
// ---------------------------------------------------------------------------
template<int ACT>
__global__ __launch_bounds__(256, 1)
void gemm_bias_act(const bf16_t* __restrict__ A,
                   const bf16_t* __restrict__ W,
                   const float*  __restrict__ bias,
                   float*  __restrict__ outF,
                   bf16_t* __restrict__ outB,
                   int M, int K, int Nout)
{
    const int lane = threadIdx.x & 31;
    const int wave = threadIdx.x >> 5;
    const int gw   = blockIdx.x * 8 + wave;
    const int tilesM = M >> 4;
    const int tileM = gw % tilesM;
    const int tileN = gw / tilesM;
    if (tileN * 16 >= Nout) return;           // wave-uniform -> EXEC stays full

    const int m0 = tileM << 4, n0 = tileN << 4;
    const int r15  = lane & 15;
    const int half = lane >> 4;

    const bf16_t* Arow = A + (size_t)(m0 + r15) * K;   // A row (M index)
    const bf16_t* Wrow = W + (size_t)(n0 + r15) * K;   // B column (N index)

    v8f acc = {};
    Frag a_cur, b_cur, a_nxt, b_nxt;
    // prologue: fragment for k0 = 0
    a_cur.h[0] = *(const v8bf*)(Arow + half * 8);
    a_cur.h[1] = *(const v8bf*)(Arow + 16 + half * 8);
    b_cur.h[0] = *(const v8bf*)(Wrow + half * 8);
    b_cur.h[1] = *(const v8bf*)(Wrow + 16 + half * 8);
    for (int k0 = 0; k0 < K; k0 += 32) {
        const int kn = k0 + 32;
        if (kn < K) {   // issue next fragment's loads before this WMMA
            a_nxt.h[0] = *(const v8bf*)(Arow + kn + half * 8);
            a_nxt.h[1] = *(const v8bf*)(Arow + kn + 16 + half * 8);
            b_nxt.h[0] = *(const v8bf*)(Wrow + kn + half * 8);
            b_nxt.h[1] = *(const v8bf*)(Wrow + kn + 16 + half * 8);
            __builtin_prefetch(Wrow + kn + 32, 0, 1);
        }
        acc = __builtin_amdgcn_wmma_f32_16x16x32_bf16(
                false, a_cur.v, false, b_cur.v, (short)0, acc, false, false);
        a_cur = a_nxt;
        b_cur = b_nxt;
    }

    const int n  = n0 + r15;
    const float bn = bias ? bias[n] : 0.f;
#pragma unroll
    for (int r = 0; r < 8; ++r) {
        const int m = m0 + (half ? r + 8 : r);
        float v = acc[r] + bn;
        if      (ACT == 1) v = fmaxf(v, 0.f);
        else if (ACT == 2) v = 1.f / (1.f + __expf(-v));
        else if (ACT == 3) v = (v > 20.f) ? v : log1pf(__expf(v));
        outF[(size_t)m * Nout + n] = v;
        if (outB) outB[(size_t)m * Nout + n] = (bf16_t)v;
    }
}

// ---------------------------------------------------------------------------
// Fused generation loss: per 16-row tile, A fragments stay in VGPRs,
// sweep V in 64-col chunks (4 waves x 16 cols), online log-sum-exp in LDS.
// All 16 B loads issued as one clause so WMMAs overlap outstanding loads.
// ---------------------------------------------------------------------------
__global__ __launch_bounds__(256, 1)
void gen_loss_tiles(const bf16_t* __restrict__ Adm,  // [RTOT, XD]
                    const bf16_t* __restrict__ Wg,   // [VSZ, XD]
                    const float*  __restrict__ gb,   // [VSZ]
                    float* __restrict__ lse)         // [RTOT]
{
    __shared__ float tile[16][65];
    __shared__ float redm[16][16];
    __shared__ float reds[16][16];

    const int lane = threadIdx.x & 31;
    const int wave = threadIdx.x >> 5;
    const int r0   = blockIdx.x << 4;
    const int r15  = lane & 15;
    const int half = lane >> 4;

    // Load the 16x256 bf16 A slab once into VGPRs (8 K-fragments).
    Frag a[8];
    const bf16_t* Arow = Adm + (size_t)(r0 + r15) * XD;
#pragma unroll
    for (int kk = 0; kk < 8; ++kk) {
        a[kk].h[0] = *(const v8bf*)(Arow + kk * 32 + half * 8);
        a[kk].h[1] = *(const v8bf*)(Arow + kk * 32 + 16 + half * 8);
    }

    // online lse state: thread -> (row rr, 4 columns starting at cc*4)
    const int rr = threadIdx.x >> 4;   // 0..15
    const int cc = threadIdx.x & 15;   // 0..15
    float m_run = -3.4e38f, s_run = 0.f;

    for (int c0 = 0; c0 < VSZ; c0 += 64) {
        const int n0 = c0 + wave * 16;            // wave-uniform
        if (n0 < VSZ) {                            // VSZ % 16 == 0 -> full tiles
            const bf16_t* Wrow = Wg + (size_t)(n0 + r15) * XD;
            // batch all 16 loads -> single clause, progressive waits
            Frag b[8];
#pragma unroll
            for (int kk = 0; kk < 8; ++kk) {
                b[kk].h[0] = *(const v8bf*)(Wrow + kk * 32 + half * 8);
                b[kk].h[1] = *(const v8bf*)(Wrow + kk * 32 + 16 + half * 8);
            }
            // prefetch next chunk of gen_w while this tile computes
            if (n0 + 64 < VSZ)
                __builtin_prefetch(Wg + (size_t)(n0 + 64 + r15) * XD, 0, 1);
            v8f acc = {};
#pragma unroll
            for (int kk = 0; kk < 8; ++kk)
                acc = __builtin_amdgcn_wmma_f32_16x16x32_bf16(
                        false, a[kk].v, false, b[kk].v, (short)0, acc, false, false);
            const float bn = gb[n0 + r15];
#pragma unroll
            for (int r = 0; r < 8; ++r)
                tile[half ? r + 8 : r][wave * 16 + r15] = acc[r] + bn;
        } else {
#pragma unroll
            for (int r = 0; r < 8; ++r)
                tile[half ? r + 8 : r][wave * 16 + r15] = -3.4e38f;
        }
        __syncthreads();
#pragma unroll
        for (int j = 0; j < 4; ++j) {
            float v = tile[rr][cc * 4 + j];
            if (v > -3.0e38f) {
                if (v > m_run) { s_run = s_run * __expf(m_run - v) + 1.f; m_run = v; }
                else           { s_run += __expf(v - m_run); }
            }
        }
        __syncthreads();
    }

    redm[rr][cc] = m_run;
    reds[rr][cc] = s_run;
    __syncthreads();
    if (cc == 0) {
        float M = -3.4e38f, S = 0.f;
#pragma unroll
        for (int j = 0; j < 16; ++j) {
            float mj = redm[rr][j], sj = reds[rr][j];
            if (sj > 0.f) {
                if (mj > M) { S = S * __expf(M - mj) + sj; M = mj; }
                else        { S += sj * __expf(mj - M); }
            }
        }
        lse[r0 + rr] = M + __logf(S);
    }
}

// target logit per row (tiny: 16384 x 256 dot products)
__global__ void tgt_logit_kernel(const bf16_t* __restrict__ Adm,
                                 const bf16_t* __restrict__ Wg,
                                 const float*  __restrict__ gb,
                                 const int*    __restrict__ ts,
                                 float* __restrict__ tgt)
{
    int i = blockIdx.x * blockDim.x + threadIdx.x;
    if (i >= RTOT) return;
    const int t = i / NB, n = i % NB;
    const int tv = ts[n * TT + t];
    const bf16_t* ar = Adm + (size_t)i * XD;
    const bf16_t* wr = Wg + (size_t)tv * XD;
    float s = 0.f;
    for (int k = 0; k < XD; ++k) s += (float)ar[k] * (float)wr[k];
    tgt[i] = s + gb[tv];
}

__global__ void gen_reduce_kernel(const float* __restrict__ lse,
                                  const float* __restrict__ tgt,
                                  float* __restrict__ out)
{
    __shared__ float sh[256];
    float s = 0.f;
    for (int i = threadIdx.x; i < RTOT; i += 256) s += lse[i] - tgt[i];
    sh[threadIdx.x] = s;
    __syncthreads();
    for (int st = 128; st > 0; st >>= 1) {
        if ((int)threadIdx.x < st) sh[threadIdx.x] += sh[threadIdx.x + st];
        __syncthreads();
    }
    if (threadIdx.x == 0) out[0] = sh[0] / (float)RTOT;
}

__global__ void cls_loss_kernel(const float* __restrict__ h,
                                const float* __restrict__ cw,
                                const float* __restrict__ cb,
                                const int*   __restrict__ labels,
                                float* __restrict__ out)
{
    __shared__ float part[NB];
    const int n = threadIdx.x;      // 64 threads
    float lg[KCLS];
#pragma unroll
    for (int k = 0; k < KCLS; ++k) {
        float s = cb[k];
        for (int j = 0; j < HD; ++j) s += h[n * HD + j] * cw[k * HD + j];
        lg[k] = s;
    }
    float M = fmaxf(fmaxf(lg[0], lg[1]), fmaxf(lg[2], lg[3]));
    float S = 0.f;
#pragma unroll
    for (int k = 0; k < KCLS; ++k) S += __expf(lg[k] - M);
    part[n] = (M + __logf(S)) - lg[labels[n]];
    __syncthreads();
    if (n == 0) {
        float s = 0.f;
        for (int j = 0; j < NB; ++j) s += part[j];
        out[1] = s / (float)NB;
    }
}

// ---------------------------------------------------------------------------
// elementwise helpers
// ---------------------------------------------------------------------------
__global__ void f2bf_kernel(bf16_t* __restrict__ dst, const float* __restrict__ src, int n) {
    int i = blockIdx.x * blockDim.x + threadIdx.x;
    if (i < n) dst[i] = (bf16_t)src[i];
}

__global__ void gather_embed_kernel(bf16_t* __restrict__ xe,
                                    const int* __restrict__ xs,
                                    const float* __restrict__ embed)
{
    size_t i = (size_t)blockIdx.x * blockDim.x + threadIdx.x;
    if (i >= (size_t)TT * NB * XD) return;
    const int x = i & (XD - 1);
    const int n = (i >> 8) & (NB - 1);
    const int t = (int)(i >> 14);
    const int id = xs[n * TT + t];
    xe[i] = (bf16_t)embed[(size_t)id * XD + x];
}

__global__ void hinit_kernel(float* __restrict__ h, bf16_t* __restrict__ hb,
                             const int* __restrict__ labels, const int* __restrict__ w,
                             const float* __restrict__ hw, const float* __restrict__ hbias,
                             const float* __restrict__ hnoise)
{
    int i = blockIdx.x * blockDim.x + threadIdx.x;
    if (i >= NB * HD) return;
    const int n = i & (HD - 1), m = i >> 9;
    float v = (float)labels[m] * hw[n] + hbias[n];
    if (*w) v += hnoise[i];
    h[i] = v; hb[i] = (bf16_t)v;
}

__global__ void pack2_kernel(bf16_t* __restrict__ dst,
                             const bf16_t* __restrict__ s1, int c1,
                             const bf16_t* __restrict__ s2, int c2, int rows)
{
    const int C = c1 + c2;
    int i = blockIdx.x * blockDim.x + threadIdx.x;
    if (i >= rows * C) return;
    const int c = i % C, r = i / C;
    dst[i] = (c < c1) ? s1[r * c1 + c] : s2[r * c2 + (c - c1)];
}

__global__ void reparam_kernel(bf16_t* __restrict__ z, const float* __restrict__ mu,
                               const float* __restrict__ stdv,
                               const float* __restrict__ eps, int t)
{
    int i = blockIdx.x * blockDim.x + threadIdx.x;
    if (i >= NB * ZD) return;
    const int zc = i & (ZD - 1), n = i >> 7;
    z[i] = (bf16_t)(eps[((size_t)n * TT + t) * ZD + zc] * stdv[i] + mu[i]);
}

__global__ void gru_combine_kernel(float* __restrict__ h, bf16_t* __restrict__ hb,
                                   const float* __restrict__ gi,
                                   const float* __restrict__ gh)
{
    int i = blockIdx.x * blockDim.x + threadIdx.x;
    if (i >= NB * HD) return;
    const int n = i & (HD - 1), m = i >> 9;
    const float* gir = gi + (size_t)m * (3 * HD);
    const float* ghr = gh + (size_t)m * (3 * HD);
    const float r  = 1.f / (1.f + __expf(-(gir[n] + ghr[n])));
    const float zg = 1.f / (1.f + __expf(-(gir[HD + n] + ghr[HD + n])));
    const float nn = tanhf(gir[2 * HD + n] + r * ghr[2 * HD + n]);
    const float hnew = (1.f - zg) * nn + zg * h[i];
    h[i] = hnew; hb[i] = (bf16_t)hnew;
}

// ---------------------------------------------------------------------------
// host side
// ---------------------------------------------------------------------------
static void launch_gemm(int act, const bf16_t* A, const bf16_t* W, const float* b,
                        float* oF, bf16_t* oB, int M, int K, int N, hipStream_t s)
{
    const int tiles  = (M / 16) * (N / 16);
    const int blocks = (tiles + 7) / 8;
    switch (act) {
    case 0: gemm_bias_act<0><<<blocks, 256, 0, s>>>(A, W, b, oF, oB, M, K, N); break;
    case 1: gemm_bias_act<1><<<blocks, 256, 0, s>>>(A, W, b, oF, oB, M, K, N); break;
    case 2: gemm_bias_act<2><<<blocks, 256, 0, s>>>(A, W, b, oF, oB, M, K, N); break;
    default: gemm_bias_act<3><<<blocks, 256, 0, s>>>(A, W, b, oF, oB, M, K, N); break;
    }
}

extern "C" void kernel_launch(void* const* d_in, const int* in_sizes, int n_in,
                              void* d_out, int out_size, void* d_ws, size_t ws_size,
                              hipStream_t stream)
{
    // --- inputs (setup_inputs dict order) ---
    const int*   xs       = (const int*)  d_in[0];
    const int*   ts       = (const int*)  d_in[1];
    const int*   labels   = (const int*)  d_in[2];
    const int*   wflag    = (const int*)  d_in[3];
    const float* eps      = (const float*)d_in[4];
    const float* h_noise  = (const float*)d_in[5];
    const float* embed    = (const float*)d_in[6];
    const float* enc_w1   = (const float*)d_in[7];
    const float* enc_b1   = (const float*)d_in[8];
    const float* enc_w2   = (const float*)d_in[9];
    const float* enc_b2   = (const float*)d_in[10];
    const float* enc_mw   = (const float*)d_in[11];
    const float* enc_mb   = (const float*)d_in[12];
    const float* enc_sw   = (const float*)d_in[13];
    const float* enc_sb   = (const float*)d_in[14];
    const float* dec_w1   = (const float*)d_in[15];
    const float* dec_b1   = (const float*)d_in[16];
    const float* dec_w2   = (const float*)d_in[17];
    const float* dec_b2   = (const float*)d_in[18];
    const float* dec_mw   = (const float*)d_in[19];
    const float* dec_mb   = (const float*)d_in[20];
    const float* gru_wih  = (const float*)d_in[21];
    const float* gru_whh  = (const float*)d_in[22];
    const float* gru_bih  = (const float*)d_in[23];
    const float* gru_bhh  = (const float*)d_in[24];
    const float* gen_w    = (const float*)d_in[25];
    const float* gen_b    = (const float*)d_in[26];
    const float* cls_w    = (const float*)d_in[27];
    const float* cls_b    = (const float*)d_in[28];
    const float* hinit_w  = (const float*)d_in[29];
    const float* hinit_b  = (const float*)d_in[30];
    float* out = (float*)d_out;

    // --- workspace carve-up (256B aligned) ---
    char* base = (char*)d_ws;
    size_t off = 0;
    auto alloc = [&](size_t bytes) {
        void* p = base + off;
        off = (off + bytes + 255) & ~(size_t)255;
        return p;
    };
    bf16_t* wEnc1 = (bf16_t*)alloc((size_t)HD * (XD + HD) * 2);
    bf16_t* wEnc2 = (bf16_t*)alloc((size_t)HD * HD * 2);
    bf16_t* wMu   = (bf16_t*)alloc((size_t)ZD * HD * 2);
    bf16_t* wStd  = (bf16_t*)alloc((size_t)ZD * HD * 2);
    bf16_t* wDec1 = (bf16_t*)alloc((size_t)HD * (ZD + HD) * 2);
    bf16_t* wDec2 = (bf16_t*)alloc((size_t)HD * HD * 2);
    bf16_t* wDm   = (bf16_t*)alloc((size_t)XD * HD * 2);
    bf16_t* wGih  = (bf16_t*)alloc((size_t)3 * HD * (XD + ZD) * 2);
    bf16_t* wGhh  = (bf16_t*)alloc((size_t)3 * HD * HD * 2);
    bf16_t* wGen  = (bf16_t*)alloc((size_t)VSZ * XD * 2);
    bf16_t* xemb  = (bf16_t*)alloc((size_t)TT * NB * XD * 2);
    bf16_t* dmall = (bf16_t*)alloc((size_t)RTOT * XD * 2);
    float*  hF    = (float*) alloc((size_t)NB * HD * 4);
    bf16_t* hB    = (bf16_t*)alloc((size_t)NB * HD * 2);
    bf16_t* pxh   = (bf16_t*)alloc((size_t)NB * (XD + HD) * 2);
    bf16_t* pzh   = (bf16_t*)alloc((size_t)NB * (ZD + HD) * 2);
    bf16_t* pxz   = (bf16_t*)alloc((size_t)NB * (XD + ZD) * 2);
    float*  e1F   = (float*) alloc((size_t)NB * HD * 4);
    bf16_t* e1B   = (bf16_t*)alloc((size_t)NB * HD * 2);
    float*  e2F   = (float*) alloc((size_t)NB * HD * 4);
    bf16_t* e2B   = (bf16_t*)alloc((size_t)NB * HD * 2);
    float*  muF   = (float*) alloc((size_t)NB * ZD * 4);
    float*  stdF  = (float*) alloc((size_t)NB * ZD * 4);
    bf16_t* zB    = (bf16_t*)alloc((size_t)NB * ZD * 2);
    float*  d1F   = (float*) alloc((size_t)NB * HD * 4);
    bf16_t* d1B   = (bf16_t*)alloc((size_t)NB * HD * 2);
    float*  d2F   = (float*) alloc((size_t)NB * HD * 4);
    bf16_t* d2B   = (bf16_t*)alloc((size_t)NB * HD * 2);
    float*  dmF   = (float*) alloc((size_t)NB * XD * 4);
    float*  giF   = (float*) alloc((size_t)NB * 3 * HD * 4);
    float*  ghF   = (float*) alloc((size_t)NB * 3 * HD * 4);
    float*  lse   = (float*) alloc((size_t)RTOT * 4);
    float*  tgt   = (float*) alloc((size_t)RTOT * 4);
    (void)ws_size; (void)n_in; (void)in_sizes; (void)out_size;

    auto conv = [&](bf16_t* dst, const float* src, int n) {
        f2bf_kernel<<<(n + 255) / 256, 256, 0, stream>>>(dst, src, n);
    };
    // weight conversions (f32 -> bf16)
    conv(wEnc1, enc_w1, HD * (XD + HD));
    conv(wEnc2, enc_w2, HD * HD);
    conv(wMu,   enc_mw, ZD * HD);
    conv(wStd,  enc_sw, ZD * HD);
    conv(wDec1, dec_w1, HD * (ZD + HD));
    conv(wDec2, dec_w2, HD * HD);
    conv(wDm,   dec_mw, XD * HD);
    conv(wGih,  gru_wih, 3 * HD * (XD + ZD));
    conv(wGhh,  gru_whh, 3 * HD * HD);
    conv(wGen,  gen_w,  VSZ * XD);

    // embedding gather [T,N,XD] bf16 + hidden init
    {
        size_t tot = (size_t)TT * NB * XD;
        gather_embed_kernel<<<(int)((tot + 255) / 256), 256, 0, stream>>>(xemb, xs, embed);
        hinit_kernel<<<(NB * HD + 255) / 256, 256, 0, stream>>>(hF, hB, labels, wflag,
                                                                hinit_w, hinit_b, h_noise);
    }

    // ---- sequential VRNN scan ----
    const int EW = (NB * HD + 255) / 256;
    for (int t = 0; t < TT; ++t) {
        const bf16_t* xt = xemb + (size_t)t * NB * XD;
        // encoder
        pack2_kernel<<<(NB * (XD + HD) + 255) / 256, 256, 0, stream>>>(pxh, xt, XD, hB, HD, NB);
        launch_gemm(1, pxh, wEnc1, enc_b1, e1F, e1B, NB, XD + HD, HD, stream);
        launch_gemm(1, e1B, wEnc2, enc_b2, e2F, e2B, NB, HD, HD, stream);
        launch_gemm(0, e2B, wMu,  enc_mb, muF,  nullptr, NB, HD, ZD, stream);
        launch_gemm(3, e2B, wStd, enc_sb, stdF, nullptr, NB, HD, ZD, stream);
        reparam_kernel<<<(NB * ZD + 255) / 256, 256, 0, stream>>>(zB, muF, stdF, eps, t);
        // decoder
        pack2_kernel<<<(NB * (ZD + HD) + 255) / 256, 256, 0, stream>>>(pzh, zB, ZD, hB, HD, NB);
        launch_gemm(1, pzh, wDec1, dec_b1, d1F, d1B, NB, ZD + HD, HD, stream);
        launch_gemm(1, d1B, wDec2, dec_b2, d2F, d2B, NB, HD, HD, stream);
        launch_gemm(2, d2B, wDm, dec_mb, dmF, dmall + (size_t)t * NB * XD, NB, HD, XD, stream);
        // GRU cell
        pack2_kernel<<<(NB * (XD + ZD) + 255) / 256, 256, 0, stream>>>(pxz, xt, XD, zB, ZD, NB);
        launch_gemm(0, pxz, wGih, gru_bih, giF, nullptr, NB, XD + ZD, 3 * HD, stream);
        launch_gemm(0, hB,  wGhh, gru_bhh, ghF, nullptr, NB, HD,      3 * HD, stream);
        gru_combine_kernel<<<EW, 256, 0, stream>>>(hF, hB, giF, ghF);
    }

    // ---- generation loss: fused WMMA GEMM + online log-sum-exp ----
    gen_loss_tiles<<<RTOT / 16, 256, 0, stream>>>(dmall, wGen, gen_b, lse);
    tgt_logit_kernel<<<(RTOT + 255) / 256, 256, 0, stream>>>(dmall, wGen, gen_b, ts, tgt);
    gen_reduce_kernel<<<1, 256, 0, stream>>>(lse, tgt, out);

    // ---- classification loss on h_last ----
    cls_loss_kernel<<<1, NB, 0, stream>>>(hF, cls_w, cls_b, labels, out);
}
